// acLSTM_35811437314834
// MI455X (gfx1250) — matrix operand order repaired and checked by reading, not compiled, not run on hardware
//
#include <hip/hip_runtime.h>
#include <hip/hip_bf16.h>

#define Hd   1024
#define Fd   132
#define Bsz  32
#define LDA  48          // LDS row stride for A tile (bf16): 96B, 16B-aligned, spreads banks

typedef __attribute__((ext_vector_type(16))) __bf16 v16bf;
typedef __attribute__((ext_vector_type(8)))  __bf16 v8bf;
typedef __attribute__((ext_vector_type(8)))  float  v8f;

// ---------------- fragment loaders (CDNA5 wave32 WMMA layouts) ----------------

// A: 16x32 bf16 (MxK). lane L<16: M=L, K=0..15; lane>=16: M=L-16, K=16..31.
__device__ __forceinline__ v16bf load_a(const __bf16* As, int mtile, int lane) {
    const int m  = mtile * 16 + (lane & 15);
    const int kb = (lane >> 4) << 4;            // 0 or 16
    const __bf16* p = As + m * LDA;
    v8bf lo = *(const v8bf*)(p + kb);           // 16B aligned
    v8bf hi = *(const v8bf*)(p + kb + 8);
    return __builtin_shufflevector(lo, hi, 0,1,2,3,4,5,6,7,8,9,10,11,12,13,14,15);
}

// B: 32x16 bf16 (KxN). lane L<16: N=L, K=0..15; lane>=16: K=16..31 (contiguous in W row).
__device__ __forceinline__ v16bf load_b(const __bf16* Wrow_k0, int lane) {
    return *(const v16bf*)(Wrow_k0 + ((lane >> 4) << 4));   // 32B aligned
}

__device__ __forceinline__ v8f wmma_bf16(v16bf a, v16bf b, v8f c) {
    // (neg_a, A, neg_b, B, c_mod, C, reuse_a, reuse_b)
    return __builtin_amdgcn_wmma_f32_16x16x32_bf16(false, a, false, b, (short)0, c,
                                                   false, false);
}

// fast stage: 4 contiguous fp32 -> 4 bf16, no bounds checks
__device__ __forceinline__ void stage_fast(__bf16* __restrict__ dst,
                                           const float* __restrict__ srow, int k0) {
    const float4 v = *(const float4*)(srow + k0);   // 16B aligned for all our sources
    dst[0] = (__bf16)v.x; dst[1] = (__bf16)v.y;
    dst[2] = (__bf16)v.z; dst[3] = (__bf16)v.w;
}

// guarded stage: only for the (statically known) ragged K-tiles of layer 1
__device__ __forceinline__ void stage_guard(__bf16* __restrict__ dst,
                                            const float* __restrict__ srow,
                                            int k0, int scol, int Kmax) {
    const int k = k0 + scol;
    dst[0] = (__bf16)((k + 0 < Kmax) ? srow[k0 + 0] : 0.f);
    dst[1] = (__bf16)((k + 1 < Kmax) ? srow[k0 + 1] : 0.f);
    dst[2] = (__bf16)((k + 2 < Kmax) ? srow[k0 + 2] : 0.f);
    dst[3] = (__bf16)((k + 3 < Kmax) ? srow[k0 + 3] : 0.f);
}

// Double-buffered K-panel GEMM: acc += A(act) x B(W^T) over KPAD (KPAD % 64 == 0).
// KPAD/KMAX are compile-time: guard selection constant-folds, loop fully unrolls,
// LDS buffer index is static (addrspace(3) -> ds_load/ds_store with const offsets).
// One __syncthreads per tile: publishes the staged buffer AND retires all reads
// of the buffer about to be overwritten.
template<int KPAD, int KMAX>
__device__ __forceinline__ void gemm_phase(v8f& acc0, v8f& acc1,
    __bf16 (&As)[2][32 * LDA],
    const float* __restrict__ src, int stride,
    const __bf16* __restrict__ Wr, int lane, int tid)
{
    const int r    = tid >> 3;
    const int scol = (tid & 7) << 2;
    const float* srow = src + (size_t)r * stride + scol;
    __bf16* d0 = &As[0][r * LDA + scol];
    __bf16* d1 = &As[1][r * LDA + scol];

    __syncthreads();                 // retire prior users of As
    stage_fast(d0, srow, 0);         // tile 0 always full (KMAX >= 32)

    #pragma unroll
    for (int k0 = 0; k0 < KPAD; k0 += 64) {
        const int k1 = k0 + 32;
        const int k2 = k0 + 64;

        __syncthreads();
        v16bf b0  = load_b(Wr + k0, lane);
        v16bf a00 = load_a(As[0], 0, lane);
        v16bf a01 = load_a(As[0], 1, lane);
        if (k1 + 32 <= KMAX) stage_fast(d1, srow, k1);       // folds at compile time
        else                 stage_guard(d1, srow, k1, scol, KMAX);
        acc0 = wmma_bf16(a00, b0, acc0);
        acc1 = wmma_bf16(a01, b0, acc1);

        __syncthreads();
        v16bf b1  = load_b(Wr + k1, lane);
        v16bf a10 = load_a(As[1], 0, lane);
        v16bf a11 = load_a(As[1], 1, lane);
        if (k2 < KPAD) {                                     // folds at compile time
            if (k2 + 32 <= KMAX) stage_fast(d0, srow, k2);
            else                 stage_guard(d0, srow, k2, scol, KMAX);
        }
        acc0 = wmma_bf16(a10, b1, acc0);
        acc1 = wmma_bf16(a11, b1, acc1);
    }
}

__device__ __forceinline__ float sigm(float v) { return 1.f / (1.f + __expf(-v)); }

// ---------------- fused LSTM cell ----------------
// grid = 32 WGs (32 hidden units each), block = 256 (8 waves).
// wave = (gate 0..3, nsub 0..1); each wave: 16 gate columns x both 16-row M tiles.
template<int KXPAD, int KXMAX>
__global__ __launch_bounds__(256) void lstm_cell_kernel(
    const float* __restrict__ x, int xStride,
    const float* __restrict__ hprev,
    const __bf16* __restrict__ Wih,       // [4H, KXPAD] bf16 row-major
    const __bf16* __restrict__ Whh,       // [4H, H]     bf16 row-major
    const float* __restrict__ bih, const float* __restrict__ bhh,
    float* __restrict__ c, float* __restrict__ hout)
{
    __shared__ __bf16 As[2][32 * LDA];
    __shared__ float  Gt[4][32][32];

    const int tid  = threadIdx.x;
    const int lane = tid & 31;
    const int wave = tid >> 5;
    const int gate = wave >> 1;
    const int nsub = wave & 1;
    const int hid0 = blockIdx.x * 32;
    const int wrow = gate * Hd + hid0 + nsub * 16 + (lane & 15);  // per-lane W row

    v8f acc0 = {};   // M rows 0..15
    v8f acc1 = {};   // M rows 16..31

    // phase 1: x @ Wih^T       phase 2: hprev @ Whh^T
    gemm_phase<KXPAD, KXMAX>(acc0, acc1, As, x, xStride,
                             Wih + (size_t)wrow * KXPAD, lane, tid);
    gemm_phase<Hd, Hd>(acc0, acc1, As, hprev, Hd,
                       Whh + (size_t)wrow * Hd, lane, tid);

    // scatter D + bias into gate exchange buffer (D layout: VGPR r <-> M=r+8*(lane>=16))
    const float bias = bih[wrow] + bhh[wrow];
    const int   nl   = nsub * 16 + (lane & 15);
    const int   mb   = (lane >> 4) << 3;     // 0 or 8
    #pragma unroll
    for (int rr = 0; rr < 8; ++rr) {
        Gt[gate][mb + rr][nl]      = acc0[rr] + bias;
        Gt[gate][16 + mb + rr][nl] = acc1[rr] + bias;
    }
    __syncthreads();

    // elementwise LSTM update for this WG's 32 hidden units (32m x 32n = 1024 elems)
    for (int e = tid; e < 32 * 32; e += 256) {
        const int m = e >> 5;
        const int n = e & 31;
        const float ig = sigm(Gt[0][m][n]);
        const float fg = sigm(Gt[1][m][n]);
        const float gg = tanhf(Gt[2][m][n]);
        const float og = sigm(Gt[3][m][n]);
        const int idx = m * Hd + hid0 + n;
        const float cn = fg * c[idx] + ig * gg;
        c[idx]    = cn;
        hout[idx] = og * tanhf(cn);
    }
}

// ---------------- decoder: out[m,f] = h2[m,:] . Wdec[f,:] + b ----------------
__global__ __launch_bounds__(128) void decoder_kernel(
    const float* __restrict__ h2, const float* __restrict__ Wdec,
    const float* __restrict__ bdec, float* __restrict__ frame,
    float* __restrict__ out, int s, int Ngen)
{
    const int idx = blockIdx.x * blockDim.x + threadIdx.x;
    if (idx >= Bsz * Fd) return;
    const int m = idx / Fd;
    const int f = idx - m * Fd;
    const float4* hv = (const float4*)(h2 + (size_t)m * Hd);
    const float4* wv = (const float4*)(Wdec + (size_t)f * Hd);
    float acc = bdec[f];
    #pragma unroll 4
    for (int k = 0; k < Hd / 4; ++k) {
        const float4 a = hv[k];
        const float4 b = wv[k];
        acc += a.x * b.x + a.y * b.y + a.z * b.z + a.w * b.w;
    }
    frame[idx] = acc;
    if (out) out[((size_t)m * Ngen + s) * Fd + f] = acc;
}

// ---------------- prep kernels ----------------
__global__ void cvt_pad_kernel(const float* __restrict__ src, __bf16* __restrict__ dst,
                               int rows, int Ks, int Kd)
{
    const int total = rows * Kd;
    for (int i = blockIdx.x * blockDim.x + threadIdx.x; i < total;
         i += gridDim.x * blockDim.x) {
        const int r = i / Kd;
        const int k = i - r * Kd;
        const float v = (k < Ks) ? src[(size_t)r * Ks + k] : 0.f;
        dst[i] = (__bf16)v;
    }
}

__global__ void zero_kernel(float* __restrict__ p, int n)
{
    for (int i = blockIdx.x * blockDim.x + threadIdx.x; i < n;
         i += gridDim.x * blockDim.x)
        p[i] = 0.f;
}

// ---------------- host ----------------
extern "C" void kernel_launch(void* const* d_in, const int* in_sizes, int n_in,
                              void* d_out, int out_size, void* d_ws, size_t ws_size,
                              hipStream_t stream)
{
    const float* initial_seq = (const float*)d_in[0];
    const float* Wih1 = (const float*)d_in[2];
    const float* Whh1 = (const float*)d_in[3];
    const float* bih1 = (const float*)d_in[4];
    const float* bhh1 = (const float*)d_in[5];
    const float* Wih2 = (const float*)d_in[6];
    const float* Whh2 = (const float*)d_in[7];
    const float* bih2 = (const float*)d_in[8];
    const float* bhh2 = (const float*)d_in[9];
    const float* Wih3 = (const float*)d_in[10];
    const float* Whh3 = (const float*)d_in[11];
    const float* bih3 = (const float*)d_in[12];
    const float* bhh3 = (const float*)d_in[13];
    const float* Wdec = (const float*)d_in[14];
    const float* bdec = (const float*)d_in[15];

    const int T    = in_sizes[0] / (Bsz * Fd);      // priming frames (64)
    const int Ngen = out_size   / (Bsz * Fd);       // generated frames (128)
    constexpr int K1P = 192;                        // 132 padded to mult of 64

    // carve workspace (256B aligned blocks)
    size_t off = 0;
    char* base = (char*)d_ws;
    auto carve = [&](size_t bytes) {
        void* p = base + off;
        off += (bytes + 255) & ~(size_t)255;
        return p;
    };
    __bf16* wih1b = (__bf16*)carve((size_t)4 * Hd * K1P * 2);
    __bf16* whh1b = (__bf16*)carve((size_t)4 * Hd * Hd * 2);
    __bf16* wih2b = (__bf16*)carve((size_t)4 * Hd * Hd * 2);
    __bf16* whh2b = (__bf16*)carve((size_t)4 * Hd * Hd * 2);
    __bf16* wih3b = (__bf16*)carve((size_t)4 * Hd * Hd * 2);
    __bf16* whh3b = (__bf16*)carve((size_t)4 * Hd * Hd * 2);
    float*  states = (float*)carve((size_t)9 * Bsz * Hd * 4);  // h0[2],h1[2],h2[2],c0,c1,c2
    float*  frame  = (float*)carve((size_t)Bsz * Fd * 4);

    float* h0[2] = { states + 0 * Bsz * Hd, states + 1 * Bsz * Hd };
    float* h1[2] = { states + 2 * Bsz * Hd, states + 3 * Bsz * Hd };
    float* h2[2] = { states + 4 * Bsz * Hd, states + 5 * Bsz * Hd };
    float* c0 = states + 6 * Bsz * Hd;
    float* c1 = states + 7 * Bsz * Hd;
    float* c2 = states + 8 * Bsz * Hd;

    // prep: bf16 weight conversion + state zeroing (every call: deterministic launch)
    cvt_pad_kernel<<<1024, 256, 0, stream>>>(Wih1, wih1b, 4 * Hd, Fd, K1P);
    cvt_pad_kernel<<<2048, 256, 0, stream>>>(Whh1, whh1b, 4 * Hd, Hd, Hd);
    cvt_pad_kernel<<<2048, 256, 0, stream>>>(Wih2, wih2b, 4 * Hd, Hd, Hd);
    cvt_pad_kernel<<<2048, 256, 0, stream>>>(Whh2, whh2b, 4 * Hd, Hd, Hd);
    cvt_pad_kernel<<<2048, 256, 0, stream>>>(Wih3, wih3b, 4 * Hd, Hd, Hd);
    cvt_pad_kernel<<<2048, 256, 0, stream>>>(Whh3, whh3b, 4 * Hd, Hd, Hd);
    zero_kernel<<<512, 256, 0, stream>>>(states, 9 * Bsz * Hd);

    const int cellGrid = Hd / 32;   // 32 WGs
    const int decGrid  = (Bsz * Fd + 127) / 128;

    // priming pass (outputs discarded; state carried)
    for (int t = 0; t < T; ++t) {
        const int pi = t & 1, po = 1 - pi;
        lstm_cell_kernel<K1P, Fd><<<cellGrid, 256, 0, stream>>>(
            initial_seq + (size_t)t * Fd, T * Fd,
            h0[pi], wih1b, whh1b, bih1, bhh1, c0, h0[po]);
        lstm_cell_kernel<Hd, Hd><<<cellGrid, 256, 0, stream>>>(
            h0[po], Hd,
            h1[pi], wih2b, whh2b, bih2, bhh2, c1, h1[po]);
        lstm_cell_kernel<Hd, Hd><<<cellGrid, 256, 0, stream>>>(
            h1[po], Hd,
            h2[pi], wih3b, whh3b, bih3, bhh3, c2, h2[po]);
    }
    // decoder output of last priming step seeds autoregression
    decoder_kernel<<<decGrid, 128, 0, stream>>>(
        h2[T & 1], Wdec, bdec, frame, (float*)nullptr, 0, Ngen);

    // autoregressive generation
    for (int s = 0; s < Ngen; ++s) {
        const int ts = T + s;
        const int pi = ts & 1, po = 1 - pi;
        lstm_cell_kernel<K1P, Fd><<<cellGrid, 256, 0, stream>>>(
            frame, Fd,
            h0[pi], wih1b, whh1b, bih1, bhh1, c0, h0[po]);
        lstm_cell_kernel<Hd, Hd><<<cellGrid, 256, 0, stream>>>(
            h0[po], Hd,
            h1[pi], wih2b, whh2b, bih2, bhh2, c1, h1[po]);
        lstm_cell_kernel<Hd, Hd><<<cellGrid, 256, 0, stream>>>(
            h1[po], Hd,
            h2[pi], wih3b, whh3b, bih3, bhh3, c2, h2[po]);
        decoder_kernel<<<decGrid, 128, 0, stream>>>(
            h2[po], Wdec, bdec, frame, (float*)d_out, s, Ngen);
    }
}
